// SNNClassifier_28183575396488
// MI455X (gfx1250) — compile-verified, matrix operand order checked by reading, not run on hardware
//
#include <hip/hip_runtime.h>

// SNN classifier: 100 LIF timesteps, each with a (16384x256)@(256x256) spike GEMM.
// Spikes are {0,1} -> exact in f16; GEMM runs on v_wmma_f32_16x16x32_f16.
// W_hidden B-fragments are register-resident (loop-invariant over t).
// v2: double-buffered spike LDS (1 barrier/timestep) + pipelined A-fragment loads.

typedef __attribute__((ext_vector_type(16))) _Float16 v16h;
typedef __attribute__((ext_vector_type(8)))  _Float16 v8h;
typedef __attribute__((ext_vector_type(8)))  float    v8f;

namespace {
constexpr int BATCH = 16384;
constexpr int F     = 256;
constexpr int H     = 256;
constexpr int C     = 10;
constexpr int SEQ   = 100;

constexpr int ROWS_PER_BLOCK = 16;   // M tile per workgroup
constexpr int ZSTR = 264;            // halves per spike row (pad 8 -> 528B stride, bank-friendly)
constexpr int ASTR = 264;            // floats per acc row
}

union Frag16 {
  v16h v;
  v8h  h2[2];
  _Float16 h[16];
};

__global__ __launch_bounds__(256)
void snn_lif_wmma_kernel(const float* __restrict__ x,
                         const float* __restrict__ Wh,
                         const float* __restrict__ bh,
                         const float* __restrict__ Wr,
                         const float* __restrict__ br,
                         float* __restrict__ out)
{
  __shared__ _Float16 zs[2][ROWS_PER_BLOCK * ZSTR];  // double-buffered spike tile (f16)
  __shared__ float    accs[ROWS_PER_BLOCK * ASTR];   // mean-spike staging for readout

  const int tid  = threadIdx.x;
  const int lane = tid & 31;        // wave32
  const int wave = tid >> 5;        // 8 waves
  const int ln   = lane & 15;       // N / M index within tile
  const int sel  = lane >> 4;       // K-half selector
  const int rowBase = blockIdx.x * ROWS_PER_BLOCK;

  // ---------------- layer-1 ownership: thread owns 16 contiguous inputs of one row
  const int r1 = tid >> 4;          // 0..15  (row within block)
  const int c1 = (tid & 15) * 16;   // 0..240 (column chunk)
  float xv[16], v1[16];
  #pragma unroll
  for (int j = 0; j < 16; ++j) {
    xv[j] = x[(size_t)(rowBase + r1) * F + c1 + j];
    v1[j] = 0.0f;
  }

  // ---------------- register-resident W_hidden B-fragments (f32 -> f16 once)
  // wave owns hidden columns [colBase, colBase+32): 2 N-tiles x 8 K-tiles
  const int colBase = wave * 32;
  Frag16 wf[8][2];
  #pragma unroll
  for (int k = 0; k < 8; ++k) {
    #pragma unroll
    for (int n = 0; n < 2; ++n) {
      const int col = colBase + n * 16 + ln;
      const int kb  = k * 32 + sel * 16;
      const float* wp = Wh + (size_t)col * F + kb;
      #pragma unroll
      for (int j = 0; j < 16; ++j) wf[k][n].h[j] = (_Float16)wp[j];
    }
  }

  const float bias0 = bh[colBase + ln];
  const float bias1 = bh[colBase + 16 + ln];

  // layer-2 state in WMMA C/D fragment layout: [tileN][vgpr]
  float v2[2][8], i2[2][8], spk[2][8];
  #pragma unroll
  for (int n = 0; n < 2; ++n)
    #pragma unroll
    for (int vv = 0; vv < 8; ++vv) { v2[n][vv] = 0.f; i2[n][vv] = 0.f; spk[n][vv] = 0.f; }

  // i1_t = x * g_t, scalar recurrence g_{t+1} = 0.8 g_t + 1, g_0 = 0
  float g = 0.0f;

  for (int t = 0; t < SEQ; ++t) {
    _Float16* zcur = zs[t & 1];

    // ---- layer-1 LIF update + spike emission (f16 into LDS)
    Frag16 zf;
    #pragma unroll
    for (int j = 0; j < 16; ++j) {
      const float vdec = 0.9f * v1[j] + 0.1f * g * xv[j];
      const bool  zb   = vdec > 1.0f;
      v1[j]  = zb ? 0.0f : vdec;
      zf.h[j] = zb ? (_Float16)1.0f : (_Float16)0.0f;
    }
    g = 0.8f * g + 1.0f;
    *(v8h*)&zcur[r1 * ZSTR + c1]     = zf.h2[0];
    *(v8h*)&zcur[r1 * ZSTR + c1 + 8] = zf.h2[1];
    __syncthreads();
    // (no trailing barrier needed: buffer t&1 is next overwritten at t+2, and the
    //  barrier at t+1 transitively orders all reads of buffer t&1 before that.)

    // ---- spike GEMM: h(16 x 32 per wave) = z(16 x 256) @ W^T slice
    // A-fragments double-buffered: prefetch k+1 while WMMA consumes k.
    v8f acc0 = {};
    v8f acc1 = {};
    Frag16 af[2];
    const _Float16* zbase = &zcur[ln * ZSTR + sel * 16];
    af[0].h2[0] = *(const v8h*)(zbase);
    af[0].h2[1] = *(const v8h*)(zbase + 8);
    #pragma unroll
    for (int k = 0; k < 8; ++k) {
      if (k < 7) {
        const _Float16* zp = zbase + (k + 1) * 32;
        af[(k + 1) & 1].h2[0] = *(const v8h*)zp;
        af[(k + 1) & 1].h2[1] = *(const v8h*)(zp + 8);
      }
      acc0 = __builtin_amdgcn_wmma_f32_16x16x32_f16(false, af[k & 1].v, false, wf[k][0].v,
                                                    (short)0, acc0, false, false);
      acc1 = __builtin_amdgcn_wmma_f32_16x16x32_f16(false, af[k & 1].v, false, wf[k][1].v,
                                                    (short)0, acc1, false, false);
    }

    // ---- layer-2 LIF directly on WMMA fragments
    #pragma unroll
    for (int vv = 0; vv < 8; ++vv) {
      {
        const float hval = acc0[vv] + bias0;
        const float vdec = 0.9f * v2[0][vv] + 0.1f * i2[0][vv];
        const float z2   = vdec > 1.0f ? 1.0f : 0.0f;
        v2[0][vv] = (1.0f - z2) * vdec;
        i2[0][vv] = 0.8f * i2[0][vv] + hval;
        spk[0][vv] += z2;
      }
      {
        const float hval = acc1[vv] + bias1;
        const float vdec = 0.9f * v2[1][vv] + 0.1f * i2[1][vv];
        const float z2   = vdec > 1.0f ? 1.0f : 0.0f;
        v2[1][vv] = (1.0f - z2) * vdec;
        i2[1][vv] = 0.8f * i2[1][vv] + hval;
        spk[1][vv] += z2;
      }
    }
  }

  __syncthreads();   // all spike-buffer traffic done before accs reuse of LDS phase

  // ---------------- stage mean spike counts to LDS (fragment -> row-major)
  #pragma unroll
  for (int n = 0; n < 2; ++n) {
    const int col = colBase + n * 16 + ln;
    #pragma unroll
    for (int vv = 0; vv < 8; ++vv) {
      const int m = vv + 8 * sel;
      accs[m * ASTR + col] = spk[n][vv] * (1.0f / (float)SEQ);
    }
  }
  __syncthreads();

  // ---------------- readout: out[16 x 10] = mean_spikes @ W_readout^T + b
  if (tid < ROWS_PER_BLOCK * C) {
    const int m = tid / C;
    const int c = tid % C;
    float s = br[c];
    const float* wrp = Wr + (size_t)c * H;
    #pragma unroll 4
    for (int k = 0; k < H; ++k) s += accs[m * ASTR + k] * wrp[k];
    out[(size_t)(rowBase + m) * C + c] = s;
  }
}

extern "C" void kernel_launch(void* const* d_in, const int* in_sizes, int n_in,
                              void* d_out, int out_size, void* d_ws, size_t ws_size,
                              hipStream_t stream) {
  (void)in_sizes; (void)n_in; (void)d_ws; (void)ws_size; (void)out_size;
  const float* x  = (const float*)d_in[0];
  const float* Wh = (const float*)d_in[1];
  const float* bh = (const float*)d_in[2];
  const float* Wr = (const float*)d_in[3];
  const float* br = (const float*)d_in[4];
  float* out = (float*)d_out;

  dim3 grid(BATCH / ROWS_PER_BLOCK);   // 1024 blocks
  dim3 block(256);                     // 8 wave32s
  snn_lif_wmma_kernel<<<grid, block, 0, stream>>>(x, Wh, bh, Wr, br, out);
}
